// MambaStack_47751446397522
// MI455X (gfx1250) — matrix operand order repaired and checked by reading, not compiled, not run on hardware
//
#include <hip/hip_runtime.h>
#include <hip/hip_bf16.h>
#include <cstddef>

// ---------------- problem constants (match reference) ----------------
#define BB       2
#define LL       1024
#define D_MODEL  768
#define N_LAYERS 2
#define D_INNER  1536
#define D_STATE  64
#define D_CONV   4
#define DT_RANK  48
#define LN_EPS   1e-5f

typedef __attribute__((ext_vector_type(16))) _Float16 v16h;
typedef __attribute__((ext_vector_type(8)))  _Float16 v8h;
typedef __attribute__((ext_vector_type(8)))  float    v8f;

__device__ __forceinline__ float siluf(float x) {
    return x * (1.0f / (1.0f + expf(-x)));
}
__device__ __forceinline__ float softplusf(float x) {
    return (x > 20.0f) ? x : log1pf(expf(x));
}

__device__ __forceinline__ v16h cat16(v8h a, v8h b) {
    return __builtin_shufflevector(a, b, 0,1,2,3,4,5,6,7,8,9,10,11,12,13,14,15);
}

// ---------------------------------------------------------------------
// Pack f32 (rows x Kin, row-stride lds) -> f16 (rows x Kp), zero-pad K.
// ---------------------------------------------------------------------
__global__ void pack_f16_kernel(const float* __restrict__ src, int lds,
                                _Float16* __restrict__ dst,
                                int rows, int Kin, int Kp)
{
    int idx = blockIdx.x * blockDim.x + threadIdx.x;
    if (idx >= rows * Kp) return;
    int r = idx / Kp;
    int k = idx - r * Kp;
    dst[idx] = (k < Kin) ? (_Float16)src[(size_t)r * lds + k] : (_Float16)0.0f;
}

// ---------------------------------------------------------------------
// WMMA GEMM on pre-packed f16: C[M,N] (+epilogue) = Ah[M,Kp] * Wh[N,Kp]^T
// One wave32 computes a 16 x (16*WN) strip; WN WMMAs share one A fragment.
// Requirements: M%16==0, N%(16*WN)==0, Kp%64==0, buffers 16B-aligned.
//   mode 0: C = acc
//   mode 1: C = softplus(acc + bias[n])             (dt projection)
//   mode 2: C = C + acc * keep[m]                   (residual out-proj)
// ---------------------------------------------------------------------
template<int WN>
__global__ void wmma_gemm_kernel(const _Float16* __restrict__ Ah,
                                 const _Float16* __restrict__ Wh,
                                 float* __restrict__ C, int ldc,
                                 int M, int N, int Kp, int mode,
                                 const float* __restrict__ bias,
                                 const unsigned char* __restrict__ mask)
{
    const int wavesPerBlock = blockDim.x >> 5;
    const int wid  = threadIdx.x >> 5;
    const int lane = threadIdx.x & 31;

    const int tilesNW = (N >> 4) / WN;
    const int tilesM  = M >> 4;

    int tile = blockIdx.x * wavesPerBlock + wid;
    tile = __builtin_amdgcn_readfirstlane(tile);     // wave-uniform tile coords
    if (tile >= tilesM * tilesNW) return;
    const int tm = tile / tilesNW;
    const int tn = tile % tilesNW;

    const int row  = lane & 15;
    const int half = lane >> 4;

    // A 16x32 f16 layout: this lane's data = two contiguous 8-half runs at
    // k0 + 8*half and k0 + 16 + 8*half  (pre-bias pointer by 8*half).
    const _Float16* arow = Ah + (size_t)(tm * 16 + row) * Kp + 8 * half;
    // B 32x16 f16 layout: two contiguous 8-half runs at k0 + 8*half and k0 + 16 + 8*half (pre-biased).
    const _Float16* brow[WN];
    #pragma unroll
    for (int j = 0; j < WN; ++j)
        brow[j] = Wh + (size_t)((tn * WN + j) * 16 + row) * Kp + 8 * half;

    v8f acc[WN];
    #pragma unroll
    for (int j = 0; j < WN; ++j) acc[j] = (v8f){};

    // K stepped by 64: issue both chunks' loads before either chunk's WMMAs.
    for (int k0 = 0; k0 < Kp; k0 += 64) {
        __builtin_prefetch(arow + k0 + 256);         // speculative global_prefetch

        const v16h af0 = cat16(*(const v8h*)(arow + k0),
                               *(const v8h*)(arow + k0 + 16));
        const v16h af1 = cat16(*(const v8h*)(arow + k0 + 32),
                               *(const v8h*)(arow + k0 + 48));
        v16h bf0[WN], bf1[WN];
        #pragma unroll
        for (int j = 0; j < WN; ++j) {
            bf0[j] = cat16(*(const v8h*)(brow[j] + k0),
                           *(const v8h*)(brow[j] + k0 + 16));
            bf1[j] = cat16(*(const v8h*)(brow[j] + k0 + 32),
                           *(const v8h*)(brow[j] + k0 + 48));
        }
        #pragma unroll
        for (int j = 0; j < WN; ++j)
            acc[j] = __builtin_amdgcn_wmma_f32_16x16x32_f16(
                false, af0, false, bf0[j], (short)0, acc[j], false, false);
        #pragma unroll
        for (int j = 0; j < WN; ++j)
            acc[j] = __builtin_amdgcn_wmma_f32_16x16x32_f16(
                false, af1, false, bf1[j], (short)0, acc[j], false, false);
    }

    // C/D layout: VGPR r, lanes 0-15 -> M=r, lanes 16-31 -> M=r+8; N = lane&15
    #pragma unroll
    for (int j = 0; j < WN; ++j) {
        const int n = (tn * WN + j) * 16 + row;
        #pragma unroll
        for (int r = 0; r < 8; ++r) {
            const int m = tm * 16 + r + 8 * half;
            float v = acc[j][r];
            if (mode == 1) {
                v = softplusf(v + bias[n]);
            } else if (mode == 2) {
                const float kp = mask[m] ? 0.0f : 1.0f;
                v = C[(size_t)m * ldc + n] + v * kp;
            }
            C[(size_t)m * ldc + n] = v;
        }
    }
}

// ------------------- x0 = x * keep -------------------
__global__ void mask_kernel(const float* __restrict__ x,
                            const unsigned char* __restrict__ mask,
                            float* __restrict__ xo, int total)
{
    int idx = blockIdx.x * blockDim.x + threadIdx.x;
    if (idx >= total) return;
    int row = idx / D_MODEL;
    xo[idx] = mask[row] ? 0.0f : x[idx];
}

// ------- causal depthwise conv (width 4) + bias + SiLU --------
__global__ void conv_silu_kernel(const float* __restrict__ xz,
                                 const float* __restrict__ cw,
                                 const float* __restrict__ cb,
                                 float* __restrict__ xc, int total)
{
    int idx = blockIdx.x * blockDim.x + threadIdx.x;
    if (idx >= total) return;
    int d = idx % D_INNER;
    int l = (idx / D_INNER) % LL;
    int b = idx / (D_INNER * LL);
    float s = cb[d];
    #pragma unroll
    for (int j = 0; j < D_CONV; ++j) {
        int ls = l - (D_CONV - 1) + j;
        if (ls >= 0)
            s += cw[d * D_CONV + j] *
                 xz[((size_t)b * LL + ls) * (2 * D_INNER) + d];
    }
    xc[idx] = siluf(s);
}

// ------------------- selective scan -------------------
__global__ void scan_kernel(const float* __restrict__ dt,
                            const float* __restrict__ dbl,
                            const float* __restrict__ xc,
                            const float* __restrict__ A_log,
                            float* __restrict__ ys)
{
    const int b = blockIdx.y;
    const int d = blockIdx.x * blockDim.x + threadIdx.x;
    __shared__ float sB[D_STATE];
    __shared__ float sC[D_STATE];

    float Arow[D_STATE];
    float h[D_STATE];
    #pragma unroll
    for (int n = 0; n < D_STATE; ++n) {
        Arow[n] = -expf(A_log[(size_t)d * D_STATE + n]);
        h[n] = 0.0f;
    }

    for (int l = 0; l < LL; ++l) {
        const size_t rowbl = (size_t)b * LL + l;
        if (threadIdx.x < 2 * D_STATE) {
            float v = dbl[rowbl * (DT_RANK + 2 * D_STATE) + DT_RANK + threadIdx.x];
            if (threadIdx.x < D_STATE) sB[threadIdx.x] = v;
            else                       sC[threadIdx.x - D_STATE] = v;
        }
        __syncthreads();

        const float dtv = dt[rowbl * D_INNER + d];
        const float xv  = xc[rowbl * D_INNER + d];
        float acc = 0.0f;
        #pragma unroll
        for (int n = 0; n < D_STATE; ++n) {
            h[n] = expf(dtv * Arow[n]) * h[n] + dtv * sB[n] * xv;
            acc += h[n] * sC[n];
        }
        ys[rowbl * D_INNER + d] = acc;
        __syncthreads();
    }
}

// --------- y = (ys + xc*Dp) * silu(z) ----------
__global__ void gate_kernel(const float* __restrict__ ys,
                            const float* __restrict__ xc,
                            const float* __restrict__ xz,
                            const float* __restrict__ Dp,
                            float* __restrict__ y, int total)
{
    int idx = blockIdx.x * blockDim.x + threadIdx.x;
    if (idx >= total) return;
    int d = idx % D_INNER;
    size_t row = (size_t)(idx / D_INNER);
    float zv = xz[row * (2 * D_INNER) + D_INNER + d];
    y[idx] = (ys[idx] + xc[idx] * Dp[d]) * siluf(zv);
}

// --------- LayerNorm over last dim + final mask ----------
__global__ void ln_kernel(const float* __restrict__ x,
                          const float* __restrict__ g,
                          const float* __restrict__ beta,
                          const unsigned char* __restrict__ mask,
                          float* __restrict__ out)
{
    const int rowi = blockIdx.x;
    const float* xr = x + (size_t)rowi * D_MODEL;
    __shared__ float s1[256];
    __shared__ float s2[256];
    float sum = 0.0f, sq = 0.0f;
    for (int i = threadIdx.x; i < D_MODEL; i += 256) {
        float v = xr[i];
        sum += v; sq += v * v;
    }
    s1[threadIdx.x] = sum; s2[threadIdx.x] = sq;
    __syncthreads();
    for (int o = 128; o > 0; o >>= 1) {
        if (threadIdx.x < o) {
            s1[threadIdx.x] += s1[threadIdx.x + o];
            s2[threadIdx.x] += s2[threadIdx.x + o];
        }
        __syncthreads();
    }
    const float mu  = s1[0] * (1.0f / D_MODEL);
    const float var = s2[0] * (1.0f / D_MODEL) - mu * mu;
    const float inv = rsqrtf(var + LN_EPS);
    const float keep = mask[rowi] ? 0.0f : 1.0f;
    for (int i = threadIdx.x; i < D_MODEL; i += 256)
        out[(size_t)rowi * D_MODEL + i] = ((xr[i] - mu) * inv * g[i] + beta[i]) * keep;
}

// ---------------------------------------------------------------------
static inline void launch_pack(const float* src, int lds, _Float16* dst,
                               int rows, int Kin, int Kp, hipStream_t s)
{
    int total = rows * Kp;
    pack_f16_kernel<<<(total + 255) / 256, 256, 0, s>>>(src, lds, dst, rows, Kin, Kp);
}

static inline void launch_gemm(const _Float16* Ah, const _Float16* Wh,
                               float* C, int ldc, int M, int N, int Kp, int mode,
                               const float* bias, const unsigned char* mask,
                               hipStream_t s)
{
    const int wavesPerBlock = 8;                       // 256 threads = 8 wave32
    if (N % 64 == 0) {
        const int tiles  = (M / 16) * (N / 64);
        const int blocks = (tiles + wavesPerBlock - 1) / wavesPerBlock;
        wmma_gemm_kernel<4><<<blocks, 32 * wavesPerBlock, 0, s>>>(
            Ah, Wh, C, ldc, M, N, Kp, mode, bias, mask);
    } else {
        const int tiles  = (M / 16) * (N / 16);
        const int blocks = (tiles + wavesPerBlock - 1) / wavesPerBlock;
        wmma_gemm_kernel<1><<<blocks, 32 * wavesPerBlock, 0, s>>>(
            Ah, Wh, C, ldc, M, N, Kp, mode, bias, mask);
    }
}

extern "C" void kernel_launch(void* const* d_in, const int* in_sizes, int n_in,
                              void* d_out, int out_size, void* d_ws, size_t ws_size,
                              hipStream_t stream)
{
    const float*         x      = (const float*)d_in[0];
    const unsigned char* mask   = (const unsigned char*)d_in[1]; // bool, 1 byte/elt
    const float*         Win    = (const float*)d_in[2];
    const float*         conv_w = (const float*)d_in[3];
    const float*         conv_b = (const float*)d_in[4];
    const float*         Wx     = (const float*)d_in[5];
    const float*         Wdt    = (const float*)d_in[6];
    const float*         bdt    = (const float*)d_in[7];
    const float*         A_log  = (const float*)d_in[8];
    const float*         Dp     = (const float*)d_in[9];
    const float*         Wout   = (const float*)d_in[10];
    const float*         ln_g   = (const float*)d_in[11];
    const float*         ln_b   = (const float*)d_in[12];
    float* out = (float*)d_out;
    (void)in_sizes; (void)n_in; (void)out_size; (void)ws_size;

    const int ROWS = BB * LL;                 // 2048
    const int K_X  = DT_RANK + 2 * D_STATE;   // 176
    const int KDT  = 64;                      // DT_RANK padded to 64

    // ---- workspace bump allocator (256B aligned) ----
    char* wp = (char*)d_ws;
    auto alloc = [&](size_t bytes) -> char* {
        char* r = wp;
        wp += (bytes + 255) & ~(size_t)255;
        return r;
    };
    float* xbuf = (float*)alloc((size_t)ROWS * D_MODEL * 4);
    float* xz   = (float*)alloc((size_t)ROWS * 2 * D_INNER * 4);
    float* xc   = (float*)alloc((size_t)ROWS * D_INNER * 4);
    float* dbl  = (float*)alloc((size_t)ROWS * K_X * 4);
    float* dtb  = (float*)alloc((size_t)ROWS * D_INNER * 4);
    float* ysb  = (float*)alloc((size_t)ROWS * D_INNER * 4);
    float* yb   = (float*)alloc((size_t)ROWS * D_INNER * 4);

    _Float16* xh    = (_Float16*)alloc((size_t)ROWS * D_MODEL * 2);
    _Float16* winh  = (_Float16*)alloc((size_t)2 * D_INNER * D_MODEL * 2);
    _Float16* xch   = (_Float16*)alloc((size_t)ROWS * D_INNER * 2);
    _Float16* wxh   = (_Float16*)alloc((size_t)K_X * D_INNER * 2);
    _Float16* dtih  = (_Float16*)alloc((size_t)ROWS * KDT * 2);
    _Float16* wdth  = (_Float16*)alloc((size_t)D_INNER * KDT * 2);
    _Float16* ybh   = (_Float16*)alloc((size_t)ROWS * D_INNER * 2);
    _Float16* wouth = (_Float16*)alloc((size_t)D_MODEL * D_INNER * 2);

    // x0 = x * keep
    {
        int total = ROWS * D_MODEL;
        mask_kernel<<<(total + 255) / 256, 256, 0, stream>>>(x, mask, xbuf, total);
    }

    for (int l = 0; l < N_LAYERS; ++l) {
        const float* Win_l  = Win    + (size_t)l * 2 * D_INNER * D_MODEL;
        const float* cw_l   = conv_w + (size_t)l * D_INNER * D_CONV;
        const float* cb_l   = conv_b + (size_t)l * D_INNER;
        const float* Wx_l   = Wx     + (size_t)l * K_X * D_INNER;
        const float* Wdt_l  = Wdt    + (size_t)l * D_INNER * DT_RANK;
        const float* bdt_l  = bdt    + (size_t)l * D_INNER;
        const float* Alog_l = A_log  + (size_t)l * D_INNER * D_STATE;
        const float* Dp_l   = Dp     + (size_t)l * D_INNER;
        const float* Wout_l = Wout   + (size_t)l * D_MODEL * D_INNER;

        // xz = x @ Win^T
        launch_pack(xbuf, D_MODEL, xh, ROWS, D_MODEL, D_MODEL, stream);
        launch_pack(Win_l, D_MODEL, winh, 2 * D_INNER, D_MODEL, D_MODEL, stream);
        launch_gemm(xh, winh, xz, 2 * D_INNER,
                    ROWS, 2 * D_INNER, D_MODEL, 0, nullptr, nullptr, stream);

        // xc = silu(conv1d(xz[:,:,:D_INNER]) + cb)
        {
            int total = ROWS * D_INNER;
            conv_silu_kernel<<<(total + 255) / 256, 256, 0, stream>>>(xz, cw_l, cb_l, xc, total);
        }

        // dbl = xc @ Wx^T       (N = 176 -> WN=1 path)
        launch_pack(xc, D_INNER, xch, ROWS, D_INNER, D_INNER, stream);
        launch_pack(Wx_l, D_INNER, wxh, K_X, D_INNER, D_INNER, stream);
        launch_gemm(xch, wxh, dbl, K_X,
                    ROWS, K_X, D_INNER, 0, nullptr, nullptr, stream);

        // dt = softplus(dbl[:,:,:48] @ Wdt^T + bdt)   (K padded 48 -> 64)
        launch_pack(dbl, K_X, dtih, ROWS, DT_RANK, KDT, stream);
        launch_pack(Wdt_l, DT_RANK, wdth, D_INNER, DT_RANK, KDT, stream);
        launch_gemm(dtih, wdth, dtb, D_INNER,
                    ROWS, D_INNER, KDT, 1, bdt_l, nullptr, stream);

        // selective scan
        {
            dim3 grid(D_INNER / 256, BB);
            scan_kernel<<<grid, 256, 0, stream>>>(dtb, dbl, xc, Alog_l, ysb);
        }

        // y = (ys + xc*Dp) * silu(z)
        {
            int total = ROWS * D_INNER;
            gate_kernel<<<(total + 255) / 256, 256, 0, stream>>>(ysb, xc, xz, Dp_l, yb, total);
        }

        // x = x + (y @ Wout^T) * keep
        launch_pack(yb, D_INNER, ybh, ROWS, D_INNER, D_INNER, stream);
        launch_pack(Wout_l, D_INNER, wouth, D_MODEL, D_INNER, D_INNER, stream);
        launch_gemm(ybh, wouth, xbuf, D_MODEL,
                    ROWS, D_MODEL, D_INNER, 2, nullptr, mask, stream);
    }

    // final LayerNorm + mask -> d_out
    ln_kernel<<<ROWS, 256, 0, stream>>>(xbuf, ln_g, ln_b, mask, out);
}
